// sLSTM_64699387347386
// MI455X (gfx1250) — matrix operand order repaired and checked by reading, not compiled, not run on hardware
//
#include <hip/hip_runtime.h>
#include <hip/hip_bf16.h>

// ---------------------------------------------------------------------------
// sLSTM (2 layers, B=16, T=1024, D=H=1024) for MI455X / gfx1250 (wave32, WMMA)
//
//   1) fp32->bf16 conversion of inputs/weights
//   2) per layer: WMMA GEMM  x_proj = X @ W^T + b   (M=16384, N=4096, K=1024)
//   3) per layer: persistent recurrence, 64 WGs; each WG's 64x1024 R slice is
//      DMA'd into LDS ONCE by the Tensor Data Mover (with pad_enable giving a
//      bank-conflict-free 516-dword row stride) and reused for all 1024 steps.
//      One grid barrier per timestep; cell state (c,n,m) lives in registers.
//   4) head GEMM  out = h_last @ Wout^T + bout
//
// Roofline: re-reading R from HBM each step would cost 2*1024*8MB ~= 16 GB
// (~0.7 ms at 23.3 TB/s); LDS residency reduces per-step global traffic to
// h (32 KB) + x_proj (256 KB) through the 192 MB L2.
// ---------------------------------------------------------------------------

typedef __attribute__((ext_vector_type(16))) __bf16 bf16x16;
typedef __attribute__((ext_vector_type(8)))  __bf16 bf16x8;
typedef __attribute__((ext_vector_type(8)))  float  fx8;
typedef __attribute__((ext_vector_type(4))) unsigned uint32x4_;
typedef __attribute__((ext_vector_type(8))) int      int32x8_;
typedef __attribute__((ext_vector_type(4))) int      int32x4_;

#define T_SEQ   1024
#define DIM     1024      // D == H
#define GDIM    4096      // 4*H
#define BATCH   16
#define NWG_REC 64

#if defined(__AMDGCN__) && __has_builtin(__builtin_amdgcn_tensor_load_to_lds)
#define USE_TDM 1
#else
#define USE_TDM 0
#endif

// Padded LDS layout produced by TDM pad feature: +2 dwords after every 256
// dwords => row stride 516 dwords (1032 bf16) and a +4 element hop at k=512.
// Rows land 4 banks apart -> conflict-free ds_load_b128 B-fragments.
__device__ __forceinline__ int r_idx(int row, int k) {
  return row * 1032 + k + ((k >= 512) ? 4 : 0);
}

__device__ __forceinline__ bf16x16 make_frag(const __bf16* p0, const __bf16* p1) {
  bf16x8 lo = *(const bf16x8*)p0;
  bf16x8 hi = *(const bf16x8*)p1;
  return __builtin_shufflevector(lo, hi, 0,1,2,3,4,5,6,7,8,9,10,11,12,13,14,15);
}

__device__ __forceinline__ fx8 wmma_bf16(bf16x16 a, bf16x16 b, fx8 c) {
  return __builtin_amdgcn_wmma_f32_16x16x32_bf16(false, a, false, b, (short)0, c,
                                                 false, false);
}

// ------------------------------ conversion ---------------------------------
__global__ void cvt_f32_bf16(const float* __restrict__ src,
                             __bf16* __restrict__ dst, int n) {
  int i = blockIdx.x * blockDim.x + threadIdx.x;
  int stride = gridDim.x * blockDim.x;
  for (; i < n; i += stride) dst[i] = (__bf16)src[i];
}

// -------------------------- input projection GEMM --------------------------
// XP[m][g] = sum_k X[b][t][k] * Wg[g][k] + bias[g],  m = t*16 + b
__global__ void __launch_bounds__(256)
proj_gemm(const __bf16* __restrict__ X, const __bf16* __restrict__ Wg,
          const float* __restrict__ bias, float* __restrict__ XP) {
  const int lane = threadIdx.x & 31;
  const int wave = threadIdx.x >> 5;
  const int wm = wave >> 2;                  // 0..1
  const int wn = wave & 3;                   // 0..3
  const int m0 = blockIdx.y * 32 + wm * 16;  // 16-row tile (one t, all b)
  const int n0 = blockIdx.x * 256 + wn * 64; // 4 N-tiles of 16
  const int half = lane >> 4;
  const int lrow = lane & 15;

  const int m = m0 + lrow;
  const int tt = m >> 4, bb = m & 15;
  const __bf16* arow = X + ((size_t)bb * T_SEQ + tt) * DIM;

  fx8 acc[4] = {};
  for (int k0 = 0; k0 < DIM; k0 += 32) {
    const __bf16* ap = arow + k0 + half * 16;       // A: contiguous 16 K per half
    bf16x16 af = make_frag(ap, ap + 8);
#pragma unroll
    for (int nt = 0; nt < 4; ++nt) {
      const __bf16* bp = Wg + (size_t)(n0 + nt * 16 + lrow) * DIM + k0 + half * 16;
      bf16x16 bfrag = make_frag(bp, bp + 8);        // B: contiguous 16 K per half
      acc[nt] = wmma_bf16(af, bfrag, acc[nt]);
    }
  }
#pragma unroll
  for (int nt = 0; nt < 4; ++nt) {
    const int n = n0 + nt * 16 + lrow;
    const float bs = bias[n];
#pragma unroll
    for (int v = 0; v < 8; ++v) {
      const int mr = m0 + ((lane < 16) ? v : v + 8);
      XP[(size_t)mr * GDIM + n] = acc[nt][v] + bs;
    }
  }
}

// ------------------------------ grid barrier -------------------------------
__device__ __forceinline__ void grid_barrier(int* ctr, int target) {
  __syncthreads();
  if (threadIdx.x == 0) {
    __threadfence();
    __hip_atomic_fetch_add(ctr, 1, __ATOMIC_RELEASE, __HIP_MEMORY_SCOPE_AGENT);
    while (__hip_atomic_load(ctr, __ATOMIC_ACQUIRE, __HIP_MEMORY_SCOPE_AGENT) <
           target) {
      __builtin_amdgcn_s_sleep(2);
    }
  }
  __syncthreads();
}

__global__ void zero_ctr(int* c) {
  if (threadIdx.x == 0 && blockIdx.x == 0) *c = 0;
}

// ------------------------------ recurrence ---------------------------------
// 64 WGs x 128 threads (4 waves). WG owns H columns [g0, g0+16).
// Wave g computes gate g's 16x16 tile: pre = xp + h @ R^T over K=1024 (32 WMMA).
__global__ void __launch_bounds__(128)
slstm_rec(const float* __restrict__ XP,    // [T][16][4096], bias already added
          const __bf16* __restrict__ Rw,   // [4096][1024] this layer's R (bf16)
          __bf16* __restrict__ hbuf,       // [2][16][1024] double-buffered h
          __bf16* __restrict__ seq_out,    // [B][T][H] layer output (bf16)
          int* __restrict__ ctr) {
  __shared__ __bf16 Rt[64 * 1032];         // ~129 KB padded R slice
  __shared__ float preS[4 * 16 * 16];      // gate-exchange tiles

  const int tid  = threadIdx.x;
  const int lane = tid & 31;
  const int gate = tid >> 5;               // wave id == gate
  const int g0   = blockIdx.x * 16;
  const int half = lane >> 4;
  const int lcol = lane & 15;

#if USE_TDM
  {
    // Tensor Data Mover: one 2-D descriptor per gate-wave, 16 rows x 1024 cols
    // of bf16 from R[gate*1024+g0 ..][0..1023] into LDS with pad_enable
    // (pad 2 dwords per 256 dwords -> 516-dword row stride, 4 banks/row).
    const unsigned long long ga =
        (unsigned long long)(const void*)(Rw + (size_t)(gate * DIM + g0) * DIM);
    const unsigned ldsa =
        (unsigned)(unsigned long long)(const void*)&Rt[0] +
        (unsigned)(gate * 16 * 1032) * 2u;

    uint32x4_ d0;
    d0[0] = 1u;                               // count=1, is_restore=0
    d0[1] = ldsa;                             // lds_addr
    d0[2] = (unsigned)(ga & 0xffffffffull);   // global_addr[31:0]
    d0[3] = (unsigned)(ga >> 32) | (2u << 30); // global_addr[56:32] | type=2

    int32x8_ d1;
    d1[0] = (1 << 16)    // data_size = 2 bytes
          | (1 << 20)    // pad_enable
          | (7 << 22)    // pad_interval = 256 dwords
          | (1 << 25);   // pad_amount  = 2 dwords
    d1[1] = (int)(1024u << 16);  // tensor_dim0[15:0]=1024 (bits 63:48)
    d1[2] = (int)(16u << 16);    // dim0[31:16]=0 | tensor_dim1[15:0]=16
    d1[3] = (int)(1024u << 16);  // dim1[31:16]=0 | tile_dim0=1024
    d1[4] = 16;                  // tile_dim1=16 | tile_dim2=0
    d1[5] = 1024;                // tensor_dim0_stride[31:0]
    d1[6] = 0;                   // stride[47:32] | tensor_dim1_stride[15:0]
    d1[7] = 0;

    int32x4_ dz = {0, 0, 0, 0};  // groups 2/3 unused (2-D tensor)
#if __clang_major__ >= 23
    int32x8_ dz8 = {0, 0, 0, 0, 0, 0, 0, 0};
    __builtin_amdgcn_tensor_load_to_lds(d0, d1, dz, dz, dz8, 0);
#else
    __builtin_amdgcn_tensor_load_to_lds(d0, d1, dz, dz, 0);
#endif
  }
#else
  // Fallback: manual staged copy into the same padded layout.
  for (int c = tid; c < 64 * 128; c += 128) {
    const int row = c >> 7, cc = c & 127;
    const int grow = (row >> 4) * DIM + g0 + (row & 15);
    const uint4* src = (const uint4*)(Rw + (size_t)grow * DIM) + cc;
    *(uint4*)&Rt[r_idx(row, cc * 8)] = *src;
  }
#endif

  // Zero initial h (own columns, buffer 0) while the DMA runs.
  for (int e = tid; e < 256; e += 128) {
    hbuf[(size_t)(e >> 4) * DIM + g0 + (e & 15)] = (__bf16)0.f;
  }

#if USE_TDM
  __builtin_amdgcn_s_wait_tensorcnt(0);  // wave consumes only its own tile
#endif
  __syncthreads();
  grid_barrier(ctr, NWG_REC);            // phase 0: h==0 visible everywhere

  // Per-thread recurrent state: 2 (b,col) elements per thread.
  const int e0 = tid * 2;
  const int sb = e0 >> 4;
  const int sc0 = e0 & 15;
  float c_[2] = {0.f, 0.f}, n_[2] = {1.f, 1.f}, mst[2] = {0.f, 0.f};

  for (int t = 0; t < T_SEQ; ++t) {
    const __bf16* hb = hbuf + (size_t)(t & 1) * BATCH * DIM;
    const float* xpt = XP + (size_t)t * BATCH * GDIM;

    // Prefetch next step's x_proj slab (global_prefetch_b8) under the WMMAs.
    if (t + 1 < T_SEQ && tid < 64) {
      const float* nxt =
          XP + (size_t)(t + 1) * BATCH * GDIM +
          (size_t)(tid & 15) * GDIM + (size_t)(tid >> 4) * DIM + g0;
      __builtin_prefetch(nxt, 0, 1);
    }

    // Seed accumulator with x_proj (+bias) in the C layout.
    fx8 acc;
#pragma unroll
    for (int v = 0; v < 8; ++v) {
      const int br = (lane < 16) ? v : v + 8;
      acc[v] = xpt[(size_t)br * GDIM + gate * DIM + g0 + lcol];
    }

    // pre_tile += h[16,K] @ R_tile^T  (A from global h, B from padded LDS R)
    const __bf16* arow = hb + (size_t)lcol * DIM;
    for (int k0 = 0; k0 < DIM; k0 += 32) {
      const __bf16* ap = arow + k0 + half * 16;
      bf16x16 af = make_frag(ap, ap + 8);
      const int kk = k0 + half * 16;
      const __bf16* bp = &Rt[r_idx(gate * 16 + lcol, kk)];
      bf16x16 bfrag = make_frag(bp, bp + 8);
      acc = wmma_bf16(af, bfrag, acc);
    }

    // Exchange gate tiles through LDS.
#pragma unroll
    for (int v = 0; v < 8; ++v) {
      const int br = (lane < 16) ? v : v + 8;
      preS[gate * 256 + br * 16 + lcol] = acc[v];
    }
    __syncthreads();

    // Elementwise sLSTM cell update (2 elements/thread, state in registers).
    __bf16* hout = hbuf + (size_t)((t + 1) & 1) * BATCH * DIM;
#pragma unroll
    for (int u = 0; u < 2; ++u) {
      const int cc = sc0 + u;
      const float zt = preS[0 * 256 + sb * 16 + cc];
      const float it = preS[1 * 256 + sb * 16 + cc];
      const float ft = preS[2 * 256 + sb * 16 + cc];
      const float ot = preS[3 * 256 + sb * 16 + cc];
      const float z = tanhf(zt);
      const float o = 1.f / (1.f + __expf(-ot));
      const float mn = fmaxf(ft + mst[u], it);
      const float ip = __expf(it - mn);
      const float fp = __expf(ft + mst[u] - mn);
      const float cn = fp * c_[u] + ip * z;
      const float nn = fp * n_[u] + ip;
      const float h = o * (cn / nn);
      c_[u] = cn; n_[u] = nn; mst[u] = mn;
      const __bf16 hb16 = (__bf16)h;
      hout[(size_t)sb * DIM + g0 + cc] = hb16;
      seq_out[((size_t)sb * T_SEQ + t) * DIM + g0 + cc] = hb16;
    }
    grid_barrier(ctr, NWG_REC * (t + 2));  // h[(t+1)&1] complete everywhere
  }
}

// ------------------------------- head GEMM ---------------------------------
// out[b][n] = sum_k Hseq[b][T-1][k] * Wo[n][k] + bo[n]   (M=16,N=1024,K=1024)
__global__ void __launch_bounds__(256)
head_gemm(const __bf16* __restrict__ Hseq, const __bf16* __restrict__ Wo,
          const float* __restrict__ bo, float* __restrict__ out) {
  const int lane = threadIdx.x & 31;
  const int wave = threadIdx.x >> 5;
  const int n0 = blockIdx.x * 512 + wave * 64;
  const int half = lane >> 4;
  const int lrow = lane & 15;
  const __bf16* arow = Hseq + ((size_t)lrow * T_SEQ + (T_SEQ - 1)) * DIM;

  fx8 acc[4] = {};
  for (int k0 = 0; k0 < DIM; k0 += 32) {
    const __bf16* ap = arow + k0 + half * 16;
    bf16x16 af = make_frag(ap, ap + 8);
#pragma unroll
    for (int nt = 0; nt < 4; ++nt) {
      const __bf16* bp = Wo + (size_t)(n0 + nt * 16 + lrow) * DIM + k0 + half * 16;
      bf16x16 bfrag = make_frag(bp, bp + 8);
      acc[nt] = wmma_bf16(af, bfrag, acc[nt]);
    }
  }
#pragma unroll
  for (int nt = 0; nt < 4; ++nt) {
    const int n = n0 + nt * 16 + lrow;
    const float bs = bo[n];
#pragma unroll
    for (int v = 0; v < 8; ++v) {
      const int mr = (lane < 16) ? v : v + 8;
      out[(size_t)mr * DIM + n] = acc[nt][v] + bs;
    }
  }
}

// -------------------------------- launcher ---------------------------------
extern "C" void kernel_launch(void* const* d_in, const int* in_sizes, int n_in,
                              void* d_out, int out_size, void* d_ws,
                              size_t ws_size, hipStream_t stream) {
  const float* input_seq = (const float*)d_in[0]; // [16,1024,1024]
  const float* W    = (const float*)d_in[1];      // [2,4096,1024]
  const float* R    = (const float*)d_in[2];      // [2,4096,1024]
  const float* bias = (const float*)d_in[3];      // [2,4096]
  const float* Wout = (const float*)d_in[4];      // [1024,1024]
  const float* bout = (const float*)d_in[5];      // [1024]
  float* out = (float*)d_out;                     // [16,1024]

  // Workspace carve-up (~386 MB total).
  char* p = (char*)d_ws;
  float*  xp   = (float*)p;  p += (size_t)T_SEQ * BATCH * GDIM * 4; // 256 MB
  __bf16* xin  = (__bf16*)p; p += (size_t)BATCH * T_SEQ * DIM * 2;  // 32 MB
  __bf16* seq1 = (__bf16*)p; p += (size_t)BATCH * T_SEQ * DIM * 2;
  __bf16* seq2 = (__bf16*)p; p += (size_t)BATCH * T_SEQ * DIM * 2;
  __bf16* Wbf  = (__bf16*)p; p += (size_t)2 * GDIM * DIM * 2;       // 16 MB
  __bf16* Rbf  = (__bf16*)p; p += (size_t)2 * GDIM * DIM * 2;       // 16 MB
  __bf16* Wobf = (__bf16*)p; p += (size_t)DIM * DIM * 2;            // 2 MB
  __bf16* hbuf = (__bf16*)p; p += (size_t)2 * BATCH * DIM * 2;
  int*    ctr  = (int*)p;    p += 256;

  cvt_f32_bf16<<<4096, 256, 0, stream>>>(input_seq, xin, BATCH * T_SEQ * DIM);
  cvt_f32_bf16<<<4096, 256, 0, stream>>>(W, Wbf, 2 * GDIM * DIM);
  cvt_f32_bf16<<<4096, 256, 0, stream>>>(R, Rbf, 2 * GDIM * DIM);
  cvt_f32_bf16<<<2048, 256, 0, stream>>>(Wout, Wobf, DIM * DIM);

  const dim3 gg(GDIM / 256, (BATCH * T_SEQ) / 32);
  for (int l = 0; l < 2; ++l) {
    const __bf16* lx = (l == 0) ? xin : seq1;
    __bf16* lo = (l == 0) ? seq1 : seq2;
    proj_gemm<<<gg, 256, 0, stream>>>(lx, Wbf + (size_t)l * GDIM * DIM,
                                      bias + (size_t)l * GDIM, xp);
    zero_ctr<<<1, 64, 0, stream>>>(ctr);
    slstm_rec<<<NWG_REC, 128, 0, stream>>>(xp, Rbf + (size_t)l * GDIM * DIM,
                                           hbuf, lo, ctr);
  }
  head_gemm<<<2, 256, 0, stream>>>(seq2, Wobf, bout, out);
}
